// LSTMModule_4655744548868
// MI455X (gfx1250) — matrix operand order repaired and checked by reading, not compiled, not run on hardware
//
#include <hip/hip_runtime.h>

// ---------------------------------------------------------------------------
// LSTM (B=256, I=256, H=128, T=512) for gfx1250 / MI455X, wave32 + WMMA bf16.
//   k0: convert W_ih / W_hh f32 -> bf16 into ws
//   k1: xg[t,b,g] = x[b,:,t] . W_ih[g,:] + b_ih[g] + b_hh[g]   (bf16 WMMA,
//       f32 accumulate, bf16 output stored in WMMA-fragment-swizzled layout)
//   k2: sequential scan, one workgroup per 16 batch rows (rows independent),
//       W_hh fragments register-resident, h through 4KB LDS, c in registers,
//       final FC fused.
// ws layout: [0,256K) W_ih bf16 | [256K,384K) W_hh bf16 | [512K, +134.2MB) xg
// ---------------------------------------------------------------------------

typedef __bf16 bf16_t;
typedef __attribute__((ext_vector_type(16))) __bf16 bf16x16;
typedef __attribute__((ext_vector_type(8)))  __bf16 bf16x8;
typedef __attribute__((ext_vector_type(8)))  float  f32x8;

__device__ __forceinline__ float sig_(float x) { return 1.f / (1.f + __expf(-x)); }
__device__ __forceinline__ float th_(float x)  { return 1.f - 2.f / (__expf(2.f * x) + 1.f); }

// ---------------------------------------------------------------- kernel 0 --
__global__ __launch_bounds__(256) void lstm_cvt(const float* __restrict__ wih,
                                               const float* __restrict__ whh,
                                               bf16_t* __restrict__ wihb,
                                               bf16_t* __restrict__ whhb) {
  int idx = blockIdx.x * 256 + threadIdx.x;
  if (idx < 512 * 256) wihb[idx] = (bf16_t)wih[idx];
  if (idx < 512 * 128) whhb[idx] = (bf16_t)whh[idx];
}

// ---------------------------------------------------------------- kernel 1 --
// grid (16 b-tiles, 8 t-chunks of 64), block 256 (8 waves).
__global__ __launch_bounds__(256, 1) void lstm_proj(const float* __restrict__ x,
                                                    const bf16_t* __restrict__ Wih,
                                                    const float* __restrict__ bih,
                                                    const float* __restrict__ bhh,
                                                    bf16_t* __restrict__ xg) {
  // A-fragment staging: [t16][ks8][lane32][16 bf16] = 128 KB
  __shared__ bf16_t ldsx[16 * 8 * 32 * 16];
  const int tid   = threadIdx.x;
  const int w     = tid >> 5;
  const int lane  = tid & 31;
  const int laneN = lane & 15;
  const int khalf = (lane & 16) ? 16 : 0;
  const int bt    = blockIdx.x;
  const int b0    = bt * 16;

  for (int c = 0; c < 4; ++c) {
    const int tbase = blockIdx.y * 64 + c * 16;
    __syncthreads();  // previous chunk's reads done before we overwrite LDS
    // ---- stage x[b0:b0+16, 0:256, tbase:tbase+16] -> bf16 A-fragment layout
    for (int p = tid; p < 4096; p += 256) {
      const int b = p >> 8;
      const int i = p & 255;
      const float4* xp =
          reinterpret_cast<const float4*>(x + ((size_t)(b0 + b) * 256 + i) * 512 + tbase);
      const int ks  = i >> 5;
      const int j   = i & 31;
      const int ls  = b + ((j & 16) ? 16 : 0);
      const int pos = j & 15;
#pragma unroll
      for (int u = 0; u < 4; ++u) {
        float4 v = xp[u];
        const int tl = u * 4;
        ldsx[(((tl + 0) * 8 + ks) * 32 + ls) * 16 + pos] = (bf16_t)v.x;
        ldsx[(((tl + 1) * 8 + ks) * 32 + ls) * 16 + pos] = (bf16_t)v.y;
        ldsx[(((tl + 2) * 8 + ks) * 32 + ls) * 16 + pos] = (bf16_t)v.z;
        ldsx[(((tl + 3) * 8 + ks) * 32 + ls) * 16 + pos] = (bf16_t)v.w;
      }
    }
    __syncthreads();
    // ---- GEMM: each wave handles 2 timesteps, all 32 gate tiles
    for (int nt = 0; nt < 32; ++nt) {
      const int g = nt * 16 + laneN;
      const float bias = bih[g] + bhh[g];
      const bf16_t* wp = Wih + (size_t)g * 256 + khalf;
      bf16x16 Bf[8];
#pragma unroll
      for (int ks = 0; ks < 8; ++ks)
        Bf[ks] = *reinterpret_cast<const bf16x16*>(wp + ks * 32);
#pragma unroll
      for (int q = 0; q < 2; ++q) {
        const int tl = w * 2 + q;
        f32x8 acc;
#pragma unroll
        for (int r = 0; r < 8; ++r) acc[r] = 0.f;
#pragma unroll
        for (int ks = 0; ks < 8; ++ks) {
          bf16x16 a =
              *reinterpret_cast<const bf16x16*>(&ldsx[((tl * 8 + ks) * 32 + lane) * 16]);
          acc = __builtin_amdgcn_wmma_f32_16x16x32_bf16(false, a, false, Bf[ks],
                                                        (short)0, acc, false, false);
        }
        bf16x8 ob;
#pragma unroll
        for (int r = 0; r < 8; ++r) ob[r] = (bf16_t)(acc[r] + bias);
        bf16_t* op =
            xg + ((((size_t)bt * 512 + (tbase + tl)) * 32 + nt) * 32 + lane) * 8;
        *reinterpret_cast<bf16x8*>(op) = ob;
      }
    }
  }
}

// ---------------------------------------------------------------- kernel 2 --
// grid 16 (one block per 16 batch rows), block 256 (8 waves).
__global__ __launch_bounds__(256, 1) void lstm_scan(const bf16_t* __restrict__ xg,
                                                    const bf16_t* __restrict__ Whh,
                                                    const float* __restrict__ Wfc,
                                                    const float* __restrict__ bfc,
                                                    float* __restrict__ out) {
  __shared__ bf16_t hbuf[4 * 32 * 16];  // h in A-fragment bf16 layout, 4 KB
  __shared__ float  hf[16 * 128];       // final h in f32 for the FC epilogue
  const int tid   = threadIdx.x;
  const int w     = tid >> 5;
  const int lane  = tid & 31;
  const int laneN = lane & 15;
  const int khalf = (lane & 16) ? 16 : 0;
  const int bt    = blockIdx.x;

  for (int p = tid; p < 4 * 32 * 16; p += 256) hbuf[p] = (bf16_t)0.f;

  // Register-resident W_hh^T fragments: wave w owns gate tiles {w, w+8, w+16, w+24}
  // -> all four gates for h-columns [16w, 16w+16) land positionally aligned.
  bf16x16 Bh[4][4];
#pragma unroll
  for (int f = 0; f < 4; ++f) {
    const int g = (w + 8 * f) * 16 + laneN;
#pragma unroll
    for (int ks = 0; ks < 4; ++ks)
      Bh[f][ks] =
          *reinterpret_cast<const bf16x16*>(Whh + (size_t)g * 128 + ks * 32 + khalf);
  }

  f32x8 cst;
  float hnew[8];
#pragma unroll
  for (int r = 0; r < 8; ++r) { cst[r] = 0.f; hnew[r] = 0.f; }

  const size_t xg_bt = (size_t)bt * 512 * 32 * 256;
  __syncthreads();

  for (int t = 0; t < 512; ++t) {
    f32x8 acc[4];
#pragma unroll
    for (int f = 0; f < 4; ++f) {  // C init = xg (+biases already folded in)
      const bf16_t* xp = xg + xg_bt + ((size_t)t * 32 + (w + 8 * f)) * 256 + lane * 8;
      bf16x8 xv = *reinterpret_cast<const bf16x8*>(xp);
#pragma unroll
      for (int r = 0; r < 8; ++r) acc[f][r] = (float)xv[r];
    }
#pragma unroll
    for (int ks = 0; ks < 4; ++ks) {  // gates += h @ W_hh^T
      bf16x16 a = *reinterpret_cast<const bf16x16*>(&hbuf[(ks * 32 + lane) * 16]);
#pragma unroll
      for (int f = 0; f < 4; ++f)
        acc[f] = __builtin_amdgcn_wmma_f32_16x16x32_bf16(false, a, false, Bh[f][ks],
                                                         (short)0, acc[f], false, false);
    }
#pragma unroll
    for (int r = 0; r < 8; ++r) {  // i,f,g,o -> c,h  (f32)
      const float ig = sig_(acc[0][r]);
      const float fg = sig_(acc[1][r]);
      const float gg = th_(acc[2][r]);
      const float og = sig_(acc[3][r]);
      const float cv = fg * cst[r] + ig * gg;
      cst[r] = cv;
      hnew[r] = og * th_(cv);
    }
    __syncthreads();  // all waves finished reading old h
    {                 // scatter new h (bf16) into A-fragment layout
      const int pos = lane & 15;
      bf16_t* hb = hbuf + (w >> 1) * 512 + pos;
#pragma unroll
      for (int r = 0; r < 8; ++r) {
        const int lp = r + ((lane & 16) ? 8 : 0) + ((w & 1) ? 16 : 0);
        hb[lp * 16] = (bf16_t)hnew[r];
      }
    }
    __syncthreads();  // new h visible before next step's reads
  }

  // ---- fused FC: out[b,j] = h_T[b,:] . W_fc[j,:] + b_fc[j]
#pragma unroll
  for (int r = 0; r < 8; ++r) {
    const int m = r + ((lane & 16) ? 8 : 0);
    hf[m * 128 + (w * 16 + laneN)] = hnew[r];
  }
  __syncthreads();
  if (tid < 32) {
    const int b = tid >> 1, j = tid & 1;
    float s = bfc[j];
    for (int k = 0; k < 128; ++k) s += hf[b * 128 + k] * Wfc[j * 128 + k];
    out[(bt * 16 + b) * 2 + j] = s;
  }
}

// ------------------------------------------------------------------ launch --
extern "C" void kernel_launch(void* const* d_in, const int* in_sizes, int n_in,
                              void* d_out, int out_size, void* d_ws, size_t ws_size,
                              hipStream_t stream) {
  const float* x    = (const float*)d_in[0];
  const float* Wih  = (const float*)d_in[1];
  const float* Whh  = (const float*)d_in[2];
  const float* b_ih = (const float*)d_in[3];
  const float* b_hh = (const float*)d_in[4];
  const float* Wfc  = (const float*)d_in[5];
  const float* b_fc = (const float*)d_in[6];

  char* ws = (char*)d_ws;
  bf16_t* Wih_b = (bf16_t*)ws;                  // 256 KB
  bf16_t* Whh_b = (bf16_t*)(ws + 262144);       // 128 KB
  bf16_t* xgbuf = (bf16_t*)(ws + 524288);       // 134.2 MB (needs ws >= ~135 MB)

  lstm_cvt<<<512, 256, 0, stream>>>(Wih, Whh, Wih_b, Whh_b);
  lstm_proj<<<dim3(16, 8), 256, 0, stream>>>(x, Wih_b, b_ih, b_hh, xgbuf);
  lstm_scan<<<16, 256, 0, stream>>>(xgbuf, Whh_b, Wfc, b_fc, (float*)d_out);
}
